// AutoReg_22411139350852
// MI455X (gfx1250) — matrix operand order repaired and checked by reading, not compiled, not run on hardware
//
#include <hip/hip_runtime.h>
#include <math.h>

// ---------------- types ----------------
typedef __bf16 bf16;
typedef __attribute__((ext_vector_type(16))) __bf16    v16bf;
typedef __attribute__((ext_vector_type(8)))  float     v8f;
typedef __attribute__((ext_vector_type(4)))  unsigned  v4u;

union Frag { v4u q[2]; v16bf v; };

// sizes
#define B_N   4096
#define D_N   112
#define H_N   512
#define G3    1536      // 3*H
#define CW    536       // 3*D + NT
#define KP    544       // CW padded to multiple of 32
#define PSL   32        // padded 3K (=30) per (b,t)

__device__ __forceinline__ bf16 f2bf(float x) {
    unsigned u = __builtin_bit_cast(unsigned, x);
    unsigned r = (u + 0x7FFFu + ((u >> 16) & 1u)) >> 16;
    unsigned short s = (unsigned short)r;
    return __builtin_bit_cast(bf16, s);
}
__device__ __forceinline__ float bf2f(bf16 h) {
    unsigned short s = __builtin_bit_cast(unsigned short, h);
    unsigned u = ((unsigned)s) << 16;
    return __builtin_bit_cast(float, u);
}
__device__ __forceinline__ float sigm(float x) { return 1.0f / (1.0f + __expf(-x)); }

// ---- A-fragment loads: row = lane%16; lo lanes K 0-7/16-23, hi lanes K 8-15/24-31 ----
template <typename PT>
__device__ __forceinline__ void loadAFrags16(Frag* A, PT base, int ld, int lane) {
    int row  = lane & 15;
    int koff = (lane & 16) ? 8 : 0;
    auto p = base + row * ld + koff;
    #pragma unroll
    for (int kk = 0; kk < 16; ++kk) {
        A[kk].q[0] = *(const v4u*)(p + kk * 32);
        A[kk].q[1] = *(const v4u*)(p + kk * 32 + 16);
    }
}

// ---- full K=512 WMMA accumulation against transposed weights WT[n][k] (bf16) ----
__device__ __forceinline__ v8f gemmK512(const Frag* A, const bf16* __restrict__ WT,
                                        int nrowBase, v8f acc, int lane) {
    int nrow = nrowBase + (lane & 15);
    const bf16* wp = WT + (size_t)nrow * 512 + ((lane & 16) >> 1); // hi lanes: +8 halfs (K 8-15/24-31)
    #pragma unroll
    for (int kk = 0; kk < 16; ++kk) {
        Frag b;
        b.q[0] = *(const v4u*)(wp + kk * 32);
        b.q[1] = *(const v4u*)(wp + kk * 32 + 16);
        acc = __builtin_amdgcn_wmma_f32_16x16x32_bf16(false, A[kk].v, false, b.v,
                                                      (short)0, acc, false, false);
    }
    return acc;
}

// ---------------- weight transpose + f32->bf16 convert ----------------
// dst[n*Kpad + k] = (k<Ksrc && n<Nsrc) ? src[(rowOff+k)*srcLd + n] : 0
__global__ void kConvT(const float* __restrict__ src, bf16* __restrict__ dst,
                       int rowOff, int Ksrc, int srcLd, int Nsrc, int Npad, int Kpad) {
    int idx = blockIdx.x * blockDim.x + threadIdx.x;
    int total = Npad * Kpad;
    if (idx >= total) return;
    int n = idx / Kpad, k = idx % Kpad;
    float v = 0.f;
    if (k < Ksrc && n < Nsrc) v = src[(size_t)(rowOff + k) * srcLd + n];
    dst[idx] = f2bf(v);
}

// const = concat([c, b, m]) -> bf16, K padded 536->544
__global__ void kBuildConst(const float* __restrict__ c, const float* __restrict__ bIn,
                            const float* __restrict__ m, bf16* __restrict__ dst) {
    int idx = blockIdx.x * blockDim.x + threadIdx.x;
    int total = B_N * KP;
    if (idx >= total) return;
    int b = idx / KP, j = idx % KP;
    float v = 0.f;
    if (j < 312)       v = c[(size_t)b * 312 + j];
    else if (j < 424)  v = bIn[(size_t)b * 112 + (j - 312)];
    else if (j < 536)  v = m[(size_t)b * 112 + (j - 424)];
    dst[idx] = f2bf(v);
}

// ---------------- one-time GEMM: C[m][n] = A[m][:] . WT[n][:] + bias[n] ----------------
template <int NCH>
__global__ void kGemm(const bf16* __restrict__ A, int lda,
                      const bf16* __restrict__ WT, int ldk,
                      const float* __restrict__ bias,
                      float* __restrict__ C, int ldc) {
    int lane = threadIdx.x & 31;
    int mt = blockIdx.x, nt = blockIdx.y;
    Frag a[NCH];
    {
        int row  = mt * 16 + (lane & 15);
        int koff = (lane & 16) ? 8 : 0;
        const bf16* p = A + (size_t)row * lda + koff;
        #pragma unroll
        for (int kk = 0; kk < NCH; ++kk) {
            a[kk].q[0] = *(const v4u*)(p + kk * 32);
            a[kk].q[1] = *(const v4u*)(p + kk * 32 + 16);
        }
    }
    v8f acc = {0.f, 0.f, 0.f, 0.f, 0.f, 0.f, 0.f, 0.f};
    {
        int nrow = nt * 16 + (lane & 15);
        const bf16* wp = WT + (size_t)nrow * ldk + ((lane & 16) >> 1);
        #pragma unroll
        for (int kk = 0; kk < NCH; ++kk) {
            Frag b;
            b.q[0] = *(const v4u*)(wp + kk * 32);
            b.q[1] = *(const v4u*)(wp + kk * 32 + 16);
            acc = __builtin_amdgcn_wmma_f32_16x16x32_bf16(false, a[kk].v, false, b.v,
                                                          (short)0, acc, false, false);
        }
    }
    int ln = lane & 15;
    int hi = (lane & 16) ? 8 : 0;
    float bs = bias[nt * 16 + ln];
    #pragma unroll
    for (int i = 0; i < 8; ++i)
        C[(size_t)(mt * 16 + i + hi) * ldc + nt * 16 + ln] = acc[i] + bs;
}

// ---------------- persistent recurrent kernel: GRU + MLP head ----------------
// 2 waves/block, each wave owns 16 batch rows; h kept in LDS (bf16), t-loop inside.
__global__ __launch_bounds__(64) void kMain(
    const float* __restrict__ z, const float* __restrict__ giConst,
    const float* __restrict__ mlpConst, const float* __restrict__ wz,
    const float* __restrict__ bhh, const float* __restrict__ b1,
    const float* __restrict__ b2,
    const bf16* __restrict__ WhhT, const bf16* __restrict__ W0hT,
    const bf16* __restrict__ W1T, const bf16* __restrict__ W2T,
    float* __restrict__ ps) {
    __shared__ bf16 hL[2][16 * 512];
    __shared__ bf16 aL[2][16 * 512];
    int lane = threadIdx.x & 31;
    int w    = threadIdx.x >> 5;
    int r0   = blockIdx.x * 32 + w * 16;
    bf16* hls = hL[w];
    bf16* als = aL[w];
    for (int i = lane; i < 16 * 512; i += 32) hls[i] = f2bf(0.f);  // h0 = 0 (per-wave, no barrier needed)

    int ln = lane & 15;
    int hi = (lane & 16) ? 8 : 0;
    Frag fr[16];

    for (int t = 0; t < D_N; ++t) {
        // ---- h (old) as A fragments ----
        loadAFrags16(fr, (const bf16*)hls, 512, lane);
        float ztD[8];
        #pragma unroll
        for (int i = 0; i < 8; ++i) {
            int row = r0 + i + hi;
            ztD[i] = (t == 0) ? -1.f : z[(size_t)row * D_N + (t - 1)];
        }
        // ---- GRU gates, 16-col tiles ----
        for (int j = 0; j < 32; ++j) {
            int col = j * 16 + ln;
            v8f aR = {0.f,0.f,0.f,0.f,0.f,0.f,0.f,0.f};
            v8f aZ = {0.f,0.f,0.f,0.f,0.f,0.f,0.f,0.f};
            v8f aN = {0.f,0.f,0.f,0.f,0.f,0.f,0.f,0.f};
            aR = gemmK512(fr, WhhT,            j * 16, aR, lane);
            aZ = gemmK512(fr, WhhT,  512 + j * 16, aZ, lane);
            aN = gemmK512(fr, WhhT, 1024 + j * 16, aN, lane);
            float wzR = wz[col], wzZ = wz[512 + col], wzN = wz[1024 + col];
            float bR = bhh[col], bZ = bhh[512 + col], bN = bhh[1024 + col];
            #pragma unroll
            for (int i = 0; i < 8; ++i) {
                size_t grow = (size_t)(r0 + i + hi);
                const float* gp = giConst + grow * G3;
                float giR = gp[col]        + ztD[i] * wzR;
                float giZ = gp[512 + col]  + ztD[i] * wzZ;
                float giN = gp[1024 + col] + ztD[i] * wzN;
                float r = sigm(giR + aR[i] + bR);
                float u = sigm(giZ + aZ[i] + bZ);
                int rl = i + hi;
                float hold = bf2f(hls[rl * 512 + j * 16 + ln]);
                float n = tanhf(giN + r * (aN[i] + bN));
                float h2 = (1.f - u) * n + u * hold;
                hls[rl * 512 + j * 16 + ln] = f2bf(h2);
            }
        }
        // ---- a1 = tanh(h2 @ W0h + mlp_const) ----
        loadAFrags16(fr, (const bf16*)hls, 512, lane);
        for (int j = 0; j < 32; ++j) {
            v8f acc = {0.f,0.f,0.f,0.f,0.f,0.f,0.f,0.f};
            acc = gemmK512(fr, W0hT, j * 16, acc, lane);
            int col = j * 16 + ln;
            #pragma unroll
            for (int i = 0; i < 8; ++i) {
                size_t grow = (size_t)(r0 + i + hi);
                float v = tanhf(acc[i] + mlpConst[grow * 512 + col]);
                als[(i + hi) * 512 + col] = f2bf(v);
            }
        }
        // ---- a2 = tanh(a1 @ W1 + b1)  (a1 lives in regs; aL reused for a2) ----
        loadAFrags16(fr, (const bf16*)als, 512, lane);
        for (int j = 0; j < 32; ++j) {
            v8f acc = {0.f,0.f,0.f,0.f,0.f,0.f,0.f,0.f};
            acc = gemmK512(fr, W1T, j * 16, acc, lane);
            float bb = b1[j * 16 + ln];
            #pragma unroll
            for (int i = 0; i < 8; ++i) {
                float v = tanhf(acc[i] + bb);
                als[(i + hi) * 512 + j * 16 + ln] = f2bf(v);
            }
        }
        // ---- p = a2 @ W2 + b2 (30 cols padded to 32) ----
        loadAFrags16(fr, (const bf16*)als, 512, lane);
        #pragma unroll
        for (int j = 0; j < 2; ++j) {
            v8f acc = {0.f,0.f,0.f,0.f,0.f,0.f,0.f,0.f};
            acc = gemmK512(fr, W2T, j * 16, acc, lane);
            int col = j * 16 + ln;
            float bb = (col < 30) ? b2[col] : 0.f;
            #pragma unroll
            for (int i = 0; i < 8; ++i) {
                size_t grow = (size_t)(r0 + i + hi);
                ps[grow * (size_t)(D_N * PSL) + (size_t)t * PSL + col] = acc[i] + bb;
            }
        }
    }
}

// ---------------- mixture-density logp + sorted mask reduction ----------------
__global__ void kPost(const float* __restrict__ z, const float* __restrict__ bIn,
                      const float* __restrict__ m, const float* __restrict__ ps,
                      float* __restrict__ out) {
    int b = blockIdx.x * blockDim.x + threadIdx.x;
    if (b >= B_N) return;
    float ll[D_N];
    float q[D_N];
    const float LOGC = 0.91893853320467274178f; // 0.5*log(2*pi)
    for (int t = 0; t < D_N; ++t) {
        const float* p = ps + (size_t)b * (D_N * PSL) + (size_t)t * PSL;
        float zt = z[(size_t)b * D_N + t];
        float m0 = -1e30f, m1 = -1e30f;
        float arg[10];
        #pragma unroll
        for (int k = 0; k < 10; ++k) {
            float lg = p[k], mu = p[10 + k], ls = p[20 + k];
            float d = (zt - mu) * __expf(-ls);
            float a = -0.5f * d * d - ls - LOGC + lg;
            arg[k] = a;
            m1 = fmaxf(m1, a);
            m0 = fmaxf(m0, lg);
        }
        float s0 = 0.f, s1 = 0.f;
        #pragma unroll
        for (int k = 0; k < 10; ++k) {
            s1 += __expf(arg[k] - m1);
            s0 += __expf(p[k] - m0);
        }
        ll[t] = (m1 + __logf(s1)) - (m0 + __logf(s0));
        q[t]  = m[(size_t)b * D_N + t] * (1.f - bIn[(size_t)b * D_N + t]);
    }
    // descending insertion sort of q
    for (int i = 1; i < D_N; ++i) {
        float key = q[i];
        int j = i - 1;
        while (j >= 0 && q[j] < key) { q[j + 1] = q[j]; --j; }
        q[j + 1] = key;
    }
    float s = 0.f;
    for (int t = 0; t < D_N; ++t) s += ll[t] * q[t];
    out[b] = s;
}

// ---------------- host ----------------
extern "C" void kernel_launch(void* const* d_in, const int* in_sizes, int n_in,
                              void* d_out, int out_size, void* d_ws, size_t ws_size,
                              hipStream_t stream) {
    const float* z   = (const float*)d_in[0];
    const float* c   = (const float*)d_in[1];
    const float* bb  = (const float*)d_in[2];
    const float* m   = (const float*)d_in[3];
    const float* Wih = (const float*)d_in[4];
    const float* Whh = (const float*)d_in[5];
    const float* bih = (const float*)d_in[6];
    const float* bhh = (const float*)d_in[7];
    const float* W0  = (const float*)d_in[8];
    const float* b0  = (const float*)d_in[9];
    const float* W1  = (const float*)d_in[10];
    const float* b1  = (const float*)d_in[11];
    const float* W2  = (const float*)d_in[12];
    const float* b2  = (const float*)d_in[13];

    char* ws = (char*)d_ws;
    size_t off = 0;
    auto alloc = [&](size_t bytes) -> char* {
        char* p = ws + off;
        off += (bytes + 255) & ~(size_t)255;
        return p;
    };
    bf16* constBf = (bf16*)alloc((size_t)B_N * KP * 2);
    bf16* WihT    = (bf16*)alloc((size_t)G3 * KP * 2);
    bf16* W0cT    = (bf16*)alloc((size_t)512 * KP * 2);
    bf16* WhhT    = (bf16*)alloc((size_t)G3 * 512 * 2);
    bf16* W0hT    = (bf16*)alloc((size_t)512 * 512 * 2);
    bf16* W1T     = (bf16*)alloc((size_t)512 * 512 * 2);
    bf16* W2T     = (bf16*)alloc((size_t)32 * 512 * 2);
    float* giC    = (float*)alloc((size_t)B_N * G3 * 4);
    float* mlpC   = (float*)alloc((size_t)B_N * 512 * 4);
    float* ps     = (float*)alloc((size_t)B_N * D_N * PSL * 4);

    auto conv = [&](const float* src, bf16* dst, int rowOff, int Ksrc, int srcLd,
                    int Nsrc, int Npad, int Kpad) {
        int total = Npad * Kpad;
        kConvT<<<(total + 255) / 256, 256, 0, stream>>>(src, dst, rowOff, Ksrc, srcLd,
                                                        Nsrc, Npad, Kpad);
    };
    conv(Wih, WihT, 1,   536, 1536, 1536, 1536, 544);   // Wih[1:]^T
    conv(W0,  W0cT, 512, 536, 512,  512,  512,  544);   // W0[H:]^T
    conv(Whh, WhhT, 0,   512, 1536, 1536, 1536, 512);   // Whh^T
    conv(W0,  W0hT, 0,   512, 512,  512,  512,  512);   // W0[:H]^T
    conv(W1,  W1T,  0,   512, 512,  512,  512,  512);   // W1^T
    conv(W2,  W2T,  0,   512, 30,   30,   32,   512);   // W2^T, N padded to 32

    {
        int total = B_N * KP;
        kBuildConst<<<(total + 255) / 256, 256, 0, stream>>>(c, bb, m, constBf);
    }
    // gi_const = const @ Wih[1:] + bih : (4096 x 1536)
    kGemm<17><<<dim3(B_N / 16, G3 / 16), 32, 0, stream>>>(constBf, KP, WihT, KP, bih, giC, G3);
    // mlp_const = const @ W0[H:] + b0 : (4096 x 512)
    kGemm<17><<<dim3(B_N / 16, 512 / 16), 32, 0, stream>>>(constBf, KP, W0cT, KP, b0, mlpC, 512);

    // recurrent core: w_z = Wih row 0
    kMain<<<B_N / 32, 64, 0, stream>>>(z, giC, mlpC, Wih, bhh, b1, b2,
                                       WhhT, W0hT, W1T, W2T, ps);

    kPost<<<B_N / 256, 256, 0, stream>>>(z, bb, m, ps, (float*)d_out);
}
